// MessagePassing_49830210568742
// MI455X (gfx1250) — matrix operand order repaired and checked by reading, not compiled, not run on hardware
//
#include <hip/hip_runtime.h>

#define NNODES 50000
#define NEDGES 800000
#define HDIM   128
#define LNEPS  1e-5f

typedef __attribute__((ext_vector_type(16))) __bf16 bf16x16;
typedef __attribute__((ext_vector_type(8)))  __bf16 bf16x8;
typedef __attribute__((ext_vector_type(8)))  float  f32x8;

union BF16Frag { bf16x16 v; bf16x8 h[2]; };

#define FRAG_ELEMS 512          // 32x16 bf16 fragment
#define OFF_EW1 0
#define OFF_EW2 (OFF_EW1 + 12*8*FRAG_ELEMS)
#define OFF_EW3 (OFF_EW2 +  4*8*FRAG_ELEMS)
#define OFF_NW1 (OFF_EW3 +  4*8*FRAG_ELEMS)
#define OFF_NW2 (OFF_NW1 +  8*8*FRAG_ELEMS)
#define OFF_NW3 (OFF_NW2 +  4*8*FRAG_ELEMS)

// ---------------------------------------------------------------------------
// Weight pre-pack: W[k][n] fp32 (Ktot x 128) -> bf16 B-fragments.
// Fragment (kt,nt): 512 bf16, lane L holds 16 contiguous values =
// W[kt*32 + (L>=16?8:0) + (p&7) + (p>=8?16:0)][nt*16 + (L&15)], p = 0..15  (B-matrix K striping)
// ---------------------------------------------------------------------------
__global__ void pack_weight(const float* __restrict__ W, __bf16* __restrict__ dst, int nKT) {
    int i = blockIdx.x * 256 + threadIdx.x;
    int total = nKT * 8 * FRAG_ELEMS;
    if (i >= total) return;
    int frag   = i >> 9;
    int within = i & 511;
    int lane = within >> 4;
    int p    = within & 15;
    int kt = frag >> 3;
    int nt = frag & 7;
    int k = kt * 32 + ((lane >= 16) ? 8 : 0) + (p & 7) + ((p >= 8) ? 16 : 0);
    int n = nt * 16 + (lane & 15);
    dst[i] = (__bf16)W[(size_t)k * HDIM + n];
}

__global__ void zero_f32(float* __restrict__ p, int n) {
    int i = blockIdx.x * 256 + threadIdx.x;
    if (i < n) p[i] = 0.0f;
}

// ---------------------------------------------------------------------------
// Fragment helpers
// ---------------------------------------------------------------------------
__device__ __forceinline__ BF16Frag load_a_f32(const float* __restrict__ p0) {
    // p0 points at this lane's first group of 8 fp32; second group is +16.
    BF16Frag a;
#pragma unroll
    for (int i = 0; i < 8; ++i) a.v[i]     = (__bf16)p0[i];
#pragma unroll
    for (int i = 0; i < 8; ++i) a.v[i + 8] = (__bf16)p0[16 + i];
    return a;
}

__device__ __forceinline__ BF16Frag load_a_lds(const __bf16* lp) {
    BF16Frag a;
    a.h[0] = *(const bf16x8*)lp;
    a.h[1] = *(const bf16x8*)(lp + 16);
    return a;
}

__device__ __forceinline__ void wmma_accum(const BF16Frag& a, const __bf16* __restrict__ wbase,
                                           f32x8 acc[8]) {
#pragma unroll
    for (int nt = 0; nt < 8; ++nt) {
        BF16Frag b;
        b.h[0] = *(const bf16x8*)(wbase + nt * FRAG_ELEMS);
        b.h[1] = *(const bf16x8*)(wbase + nt * FRAG_ELEMS + 8);
        acc[nt] = __builtin_amdgcn_wmma_f32_16x16x32_bf16(
            false, a.v, false, b.v, (short)0, acc[nt], false, false);
    }
}

__device__ __forceinline__ void reset_acc(f32x8 acc[8]) {
    f32x8 z = {};
#pragma unroll
    for (int i = 0; i < 8; ++i) acc[i] = z;
}

// bias + ReLU, write bf16 row-major 16x128 (stride 136) into LDS tile.
// D layout: VGPR v of lane -> row v + hi*8, col nt*16 + (lane&15).
__device__ __forceinline__ void relu_bias_to_lds(__bf16 (*buf)[136], f32x8 acc[8],
                                                 const float* __restrict__ bias, int r, int hi) {
#pragma unroll
    for (int nt = 0; nt < 8; ++nt) {
        float b = bias[nt * 16 + r];
#pragma unroll
        for (int v = 0; v < 8; ++v) {
            float x = acc[nt][v] + b;
            buf[v + hi * 8][nt * 16 + r] = (__bf16)(x > 0.0f ? x : 0.0f);
        }
    }
}

// H->H hidden layer entirely from an LDS tile.
__device__ __forceinline__ void hidden_layer(const __bf16 (*in)[136],
                                             const __bf16* __restrict__ wf,
                                             f32x8 acc[8], int r, int kin, int lane) {
#pragma unroll
    for (int kt = 0; kt < 4; ++kt) {
        BF16Frag a = load_a_lds(&in[r][kt * 32 + kin]);
        wmma_accum(a, wf + kt * (8 * FRAG_ELEMS) + lane * 16, acc);
    }
}

// bias + LayerNorm stats on the C/D layout. Per row r_v = v + hi*8.
__device__ __forceinline__ void ln_stats(f32x8 acc[8], const float* __restrict__ bias,
                                         int r, float mean[8], float rstd[8]) {
#pragma unroll
    for (int nt = 0; nt < 8; ++nt) {
        float b = bias[nt * 16 + r];
#pragma unroll
        for (int v = 0; v < 8; ++v) acc[nt][v] += b;
    }
#pragma unroll
    for (int v = 0; v < 8; ++v) {
        float s = 0.0f, q = 0.0f;
#pragma unroll
        for (int nt = 0; nt < 8; ++nt) { float x = acc[nt][v]; s += x; q += x * x; }
#pragma unroll
        for (int m = 1; m < 16; m <<= 1) {   // reduce within 16-lane half
            s += __shfl_xor(s, m, 32);
            q += __shfl_xor(q, m, 32);
        }
        float mu  = s * (1.0f / HDIM);
        float var = q * (1.0f / HDIM) - mu * mu;
        mean[v] = mu;
        rstd[v] = rsqrtf(var + LNEPS);
    }
}

// ---------------------------------------------------------------------------
// Edge kernel: one wave32 per 16-edge tile.
// ---------------------------------------------------------------------------
__launch_bounds__(32)
__global__ void edge_kernel(const float* __restrict__ nodes, const float* __restrict__ edges,
                            const int* __restrict__ senders, const int* __restrict__ receivers,
                            const float* __restrict__ eb1, const float* __restrict__ eb2,
                            const float* __restrict__ eb3, const float* __restrict__ eg,
                            const float* __restrict__ ebt,
                            const __bf16* __restrict__ wfrags,
                            float* __restrict__ effects, float* __restrict__ edges_out) {
    __shared__ __align__(16) __bf16 lds[2][16][136];
    const int lane = threadIdx.x;
    const int r    = lane & 15;
    const int hi   = lane >> 4;
    const int kin  = hi * 8;
    const int t    = blockIdx.x;
    const int e    = t * 16 + r;

    const int recv = receivers[e];
    const int send = senders[e];
    const float* dstrow = nodes + (size_t)recv * HDIM;
    const float* srcrow = nodes + (size_t)send * HDIM;
    const float* edgrow = edges + (size_t)e * HDIM;

    f32x8 acc[8];
    reset_acc(acc);

    // Layer 1: [dst | src | edge] (K=384) @ ew1
#pragma unroll
    for (int kt = 0; kt < 12; ++kt) {
        const float* base = (kt < 4) ? dstrow : (kt < 8) ? srcrow : edgrow;
        int c0 = (kt & 3) * 32;
        BF16Frag a = load_a_f32(base + c0 + kin);
        wmma_accum(a, wfrags + OFF_EW1 + kt * (8 * FRAG_ELEMS) + lane * 16, acc);
    }
    relu_bias_to_lds(lds[0], acc, eb1, r, hi);
    __syncthreads();   // single-wave WG: S_NOP + compiler ordering

    // Layer 2
    reset_acc(acc);
    hidden_layer(lds[0], wfrags + OFF_EW2, acc, r, kin, lane);
    relu_bias_to_lds(lds[1], acc, eb2, r, hi);
    __syncthreads();

    // Layer 3 + LayerNorm
    reset_acc(acc);
    hidden_layer(lds[1], wfrags + OFF_EW3, acc, r, kin, lane);
    float mean[8], rstd[8];
    ln_stats(acc, eb3, r, mean, rstd);

    const int rowbase = t * 16 + hi * 8;
    int rv[8];
#pragma unroll
    for (int v = 0; v < 8; ++v) rv[v] = receivers[rowbase + v];

#pragma unroll
    for (int nt = 0; nt < 8; ++nt) {
        int col = nt * 16 + r;
        float g = eg[col], bb = ebt[col];
#pragma unroll
        for (int v = 0; v < 8; ++v) {
            float y = (acc[nt][v] - mean[v]) * rstd[v] * g + bb;
            edges_out[(size_t)(rowbase + v) * HDIM + col] = y;
            unsafeAtomicAdd(effects + (size_t)rv[v] * HDIM + col, y);  // global_atomic_add_f32
        }
    }
}

// ---------------------------------------------------------------------------
// Node kernel: one wave32 per 16-node tile.
// ---------------------------------------------------------------------------
__launch_bounds__(32)
__global__ void node_kernel(const float* __restrict__ nodes, const float* __restrict__ effects,
                            const float* __restrict__ nb1, const float* __restrict__ nb2,
                            const float* __restrict__ nb3, const float* __restrict__ ng,
                            const float* __restrict__ nbt,
                            const __bf16* __restrict__ wfrags,
                            float* __restrict__ nodes_out) {
    __shared__ __align__(16) __bf16 lds[2][16][136];
    const int lane = threadIdx.x;
    const int r    = lane & 15;
    const int hi   = lane >> 4;
    const int kin  = hi * 8;
    const int t    = blockIdx.x;

    const float* noderow = nodes   + (size_t)(t * 16 + r) * HDIM;
    const float* effrow  = effects + (size_t)(t * 16 + r) * HDIM;

    f32x8 acc[8];
    reset_acc(acc);

    // Layer 1: [nodes | effects] (K=256) @ nw1
#pragma unroll
    for (int kt = 0; kt < 8; ++kt) {
        const float* base = (kt < 4) ? noderow : effrow;
        int c0 = (kt & 3) * 32;
        BF16Frag a = load_a_f32(base + c0 + kin);
        wmma_accum(a, wfrags + OFF_NW1 + kt * (8 * FRAG_ELEMS) + lane * 16, acc);
    }
    relu_bias_to_lds(lds[0], acc, nb1, r, hi);
    __syncthreads();

    reset_acc(acc);
    hidden_layer(lds[0], wfrags + OFF_NW2, acc, r, kin, lane);
    relu_bias_to_lds(lds[1], acc, nb2, r, hi);
    __syncthreads();

    reset_acc(acc);
    hidden_layer(lds[1], wfrags + OFF_NW3, acc, r, kin, lane);
    float mean[8], rstd[8];
    ln_stats(acc, nb3, r, mean, rstd);

    const int rowbase = t * 16 + hi * 8;
#pragma unroll
    for (int nt = 0; nt < 8; ++nt) {
        int col = nt * 16 + r;
        float g = ng[col], bb = nbt[col];
#pragma unroll
        for (int v = 0; v < 8; ++v) {
            float y = (acc[nt][v] - mean[v]) * rstd[v] * g + bb;
            nodes_out[(size_t)(rowbase + v) * HDIM + col] = y;
        }
    }
}

// ---------------------------------------------------------------------------
extern "C" void kernel_launch(void* const* d_in, const int* in_sizes, int n_in,
                              void* d_out, int out_size, void* d_ws, size_t ws_size,
                              hipStream_t stream) {
    const float* nodes     = (const float*)d_in[0];
    const float* edges     = (const float*)d_in[1];
    const int*   senders   = (const int*)d_in[2];
    const int*   receivers = (const int*)d_in[3];
    const float* ew1 = (const float*)d_in[4];  const float* eb1 = (const float*)d_in[5];
    const float* ew2 = (const float*)d_in[6];  const float* eb2 = (const float*)d_in[7];
    const float* ew3 = (const float*)d_in[8];  const float* eb3 = (const float*)d_in[9];
    const float* eg  = (const float*)d_in[10]; const float* ebt = (const float*)d_in[11];
    const float* nw1 = (const float*)d_in[12]; const float* nb1 = (const float*)d_in[13];
    const float* nw2 = (const float*)d_in[14]; const float* nb2 = (const float*)d_in[15];
    const float* nw3 = (const float*)d_in[16]; const float* nb3 = (const float*)d_in[17];
    const float* ng  = (const float*)d_in[18]; const float* nbt = (const float*)d_in[19];

    // Workspace layout: [effects: N*H fp32][packed bf16 weight fragments]
    float*  effects = (float*)d_ws;
    __bf16* wfrags  = (__bf16*)((char*)d_ws + (size_t)NNODES * HDIM * sizeof(float));

    float* nodes_out = (float*)d_out;                          // [N, H]
    float* edges_out = (float*)d_out + (size_t)NNODES * HDIM;  // [E, H]

    // Pack weights (tiny) + zero the scatter accumulator.
    pack_weight<<<(12 * 8 * FRAG_ELEMS + 255) / 256, 256, 0, stream>>>(ew1, wfrags + OFF_EW1, 12);
    pack_weight<<<( 4 * 8 * FRAG_ELEMS + 255) / 256, 256, 0, stream>>>(ew2, wfrags + OFF_EW2, 4);
    pack_weight<<<( 4 * 8 * FRAG_ELEMS + 255) / 256, 256, 0, stream>>>(ew3, wfrags + OFF_EW3, 4);
    pack_weight<<<( 8 * 8 * FRAG_ELEMS + 255) / 256, 256, 0, stream>>>(nw1, wfrags + OFF_NW1, 8);
    pack_weight<<<( 4 * 8 * FRAG_ELEMS + 255) / 256, 256, 0, stream>>>(nw2, wfrags + OFF_NW2, 4);
    pack_weight<<<( 4 * 8 * FRAG_ELEMS + 255) / 256, 256, 0, stream>>>(nw3, wfrags + OFF_NW3, 4);
    zero_f32<<<(NNODES * HDIM + 255) / 256, 256, 0, stream>>>(effects, NNODES * HDIM);

    edge_kernel<<<NEDGES / 16, 32, 0, stream>>>(nodes, edges, senders, receivers,
                                                eb1, eb2, eb3, eg, ebt,
                                                wfrags, effects, edges_out);

    node_kernel<<<NNODES / 16, 32, 0, stream>>>(nodes, effects,
                                                nb1, nb2, nb3, ng, nbt,
                                                wfrags, nodes_out);
}
